// PostEncode_50964081934345
// MI455X (gfx1250) — compile-verified, matrix operand order checked by reading, not run on hardware
//
#include <hip/hip_runtime.h>
#include <hip/hip_bf16.h>

typedef __attribute__((ext_vector_type(16))) _Float16 v16h;
typedef __attribute__((ext_vector_type(8)))  _Float16 v8h;
typedef __attribute__((ext_vector_type(8)))  float    v8f;

#define N_L   50
#define EDIM  64
#define CEDIM 128

// ---------------------------------------------------------------------------
// Fragment helpers (wave32 WMMA 16x16x32 f16, per CDNA5 ISA VGPR layouts)
// A fragment: lane (l&15) = matrix row, group (l>>4): halves [g*8..g*8+7] and
// [g*8+16..g*8+23] of the 32-wide K chunk. B = W^T so the per-lane B fragment
// is a contiguous slice of W's row n with identical K ordering.
// ---------------------------------------------------------------------------
__device__ __forceinline__ v16h load_frag(const _Float16* p) {
  v8h lo = *(const v8h*)(p);
  v8h hi = *(const v8h*)(p + 16);
  return __builtin_shufflevector(lo, hi, 0, 1, 2, 3, 4, 5, 6, 7,
                                         8, 9, 10, 11, 12, 13, 14, 15);
}

__device__ __forceinline__ v8f wmma_f16(v16h a, v16h b, v8f c) {
  return __builtin_amdgcn_wmma_f32_16x16x32_f16(
      /*neg_a=*/false, a, /*neg_b=*/false, b,
      /*c_mod=*/(short)0, c, /*reuse_a=*/false, /*reuse_b=*/false);
}

// C/D layout: lane (l&15) = col n, VGPR v = row ((l>>4)*8 + v) of the tile.
__device__ __forceinline__ void store_frag_relu(v8f c, _Float16* Y, int ystride,
                                                int row0, int col0, int lane,
                                                float bias) {
  const int n  = col0 + (lane & 15);
  const int m0 = row0 + ((lane >> 4) << 3);
#pragma unroll
  for (int v = 0; v < 8; ++v) {
    float val = fmaxf(c[v] + bias, 0.0f);
    Y[(m0 + v) * ystride + n] = (_Float16)val;
  }
}

// Y[M=64, 64] = relu(X[M=64, KDIM] @ W[64, KDIM]^T + bias); wave w owns M-tile w.
template <int KDIM>
__device__ __forceinline__ void wmma_layer(const _Float16* __restrict__ X, int xstride,
                                           const _Float16* __restrict__ W,    // f16 [64][KDIM]
                                           const float* __restrict__ bias,    // f32 [64]
                                           _Float16* __restrict__ Y, int ystride,
                                           int wave, int lane) {
  const int row0 = wave * 16;
  const int rA   = row0 + (lane & 15);
  const int g8   = (lane >> 4) * 8;
  const int nlo  = lane & 15;
  v8f c0 = {}, c1 = {}, c2 = {}, c3 = {};
#pragma unroll
  for (int kc = 0; kc < KDIM; kc += 32) {
    v16h a  = load_frag(X + rA * xstride + kc + g8);
    v16h b0 = load_frag(W + (0 * 16 + nlo) * KDIM + kc + g8);
    v16h b1 = load_frag(W + (1 * 16 + nlo) * KDIM + kc + g8);
    v16h b2 = load_frag(W + (2 * 16 + nlo) * KDIM + kc + g8);
    v16h b3 = load_frag(W + (3 * 16 + nlo) * KDIM + kc + g8);
    c0 = wmma_f16(a, b0, c0);
    c1 = wmma_f16(a, b1, c1);
    c2 = wmma_f16(a, b2, c2);
    c3 = wmma_f16(a, b3, c3);
  }
  store_frag_relu(c0, Y, ystride, row0, 0,  lane, bias[0  + nlo]);
  store_frag_relu(c1, Y, ystride, row0, 16, lane, bias[16 + nlo]);
  store_frag_relu(c2, Y, ystride, row0, 32, lane, bias[32 + nlo]);
  store_frag_relu(c3, Y, ystride, row0, 48, lane, bias[48 + nlo]);
}

// Attention layer 1: input is concat(o[l], post_rep) along K; the second half
// of K is row-independent, so A-fragments for kc>=64 broadcast from sPostH.
__device__ __forceinline__ void wmma_att1(const _Float16* __restrict__ O,     // [64][64]
                                          const _Float16* __restrict__ postH, // [64]
                                          const _Float16* __restrict__ W,     // f16 [64][128]
                                          const float* __restrict__ bias,
                                          _Float16* __restrict__ Y,           // [64][64]
                                          int wave, int lane) {
  const int row0 = wave * 16;
  const int rA   = row0 + (lane & 15);
  const int g8   = (lane >> 4) * 8;
  const int nlo  = lane & 15;
  v8f c0 = {}, c1 = {}, c2 = {}, c3 = {};
#pragma unroll
  for (int kc = 0; kc < 128; kc += 32) {
    v16h a = (kc < 64) ? load_frag(O + rA * 64 + kc + g8)
                       : load_frag(postH + (kc - 64) + g8);
    v16h b0 = load_frag(W + (0 * 16 + nlo) * 128 + kc + g8);
    v16h b1 = load_frag(W + (1 * 16 + nlo) * 128 + kc + g8);
    v16h b2 = load_frag(W + (2 * 16 + nlo) * 128 + kc + g8);
    v16h b3 = load_frag(W + (3 * 16 + nlo) * 128 + kc + g8);
    c0 = wmma_f16(a, b0, c0);
    c1 = wmma_f16(a, b1, c1);
    c2 = wmma_f16(a, b2, c2);
    c3 = wmma_f16(a, b3, c3);
  }
  store_frag_relu(c0, Y, 64, row0, 0,  lane, bias[0  + nlo]);
  store_frag_relu(c1, Y, 64, row0, 16, lane, bias[16 + nlo]);
  store_frag_relu(c2, Y, 64, row0, 32, lane, bias[32 + nlo]);
  store_frag_relu(c3, Y, 64, row0, 48, lane, bias[48 + nlo]);
}

// ---------------------------------------------------------------------------
// Weight fp32 -> f16 conversion pre-pass
// ---------------------------------------------------------------------------
__global__ void cvt_f32_to_f16(const float* __restrict__ src,
                               _Float16* __restrict__ dst, int n) {
  int i = blockIdx.x * blockDim.x + threadIdx.x;
  if (i < n) dst[i] = (_Float16)src[i];
}

// ---------------------------------------------------------------------------
// Main kernel: one workgroup (4 wave32) per node.
// ---------------------------------------------------------------------------
__global__ void __launch_bounds__(128)
post_encode_kernel(const int* __restrict__ nodes,
                   const int* __restrict__ pu_history,
                   const int* __restrict__ pr_history,
                   const int* __restrict__ hist_len,
                   const float* __restrict__ u2e,
                   const float* __restrict__ r2e,
                   const float* __restrict__ p2e,
                   const float* __restrict__ w_e_w, const float* __restrict__ w_e_b,
                   const _Float16* __restrict__ w1h, const float* __restrict__ w_1_b,
                   const _Float16* __restrict__ w2h, const float* __restrict__ w_2_b,
                   const _Float16* __restrict__ a1h, const float* __restrict__ att1_b,
                   const _Float16* __restrict__ a2h, const float* __restrict__ att2_b,
                   const float* __restrict__ att3_w, const float* __restrict__ att3_b,
                   const float* __restrict__ o_w_w, const float* __restrict__ o_w_b,
                   float* __restrict__ out) {
  __shared__ __align__(16) _Float16 sX[64 * 128];   // concat(u,r) f16 (rows>=50 zero)
  __shared__ __align__(16) _Float16 sY[64 * 64];    // layer intermediates
  __shared__ __align__(16) _Float16 sO[64 * 64];    // o = relu(x@W2^T) kept to the end
  __shared__ __align__(16) _Float16 sPostH[64];
  __shared__ float sPost[64];
  __shared__ float sScore[64];
  __shared__ float sWt[64];
  __shared__ float sAH[64];
  __shared__ int   sPU[64];
  __shared__ int   sPR[64];

  const int n    = blockIdx.x;
  const int tid  = threadIdx.x;
  const int lane = tid & 31;
  const int wave = tid >> 5;

  // Stage per-node history indices.
  if (tid < N_L) {
    sPU[tid] = pu_history[(size_t)n * N_L + tid];
    sPR[tid] = pr_history[(size_t)n * N_L + tid];
  }
  // post_rep = relu(W_e @ p2e[node] + b_e)  (tiny mat-vec: plain f32 VALU)
  if (tid < EDIM) {
    const float4* prow = (const float4*)(p2e + (size_t)nodes[n] * CEDIM);
    const float4* wrow = (const float4*)(w_e_w + (size_t)tid * CEDIM);
    float acc = w_e_b[tid];
#pragma unroll
    for (int k = 0; k < CEDIM / 4; ++k) {
      float4 pv = prow[k], wv = wrow[k];
      acc += pv.x * wv.x + pv.y * wv.y + pv.z * wv.z + pv.w * wv.w;
    }
    acc = fmaxf(acc, 0.0f);
    sPost[tid]  = acc;
    sPostH[tid] = (_Float16)acc;
  }
  __syncthreads();

  // Gather concat(u, r) -> sX [64][128] f16; coalesced 16B gathers (L2-hot table).
  for (int base = tid * 4; base < 64 * 128; base += 128 * 4) {
    const int row = base >> 7, col = base & 127;
    float4 v = make_float4(0.f, 0.f, 0.f, 0.f);
    if (row < N_L) {
      if (col < EDIM) v = *(const float4*)(u2e + (size_t)sPU[row] * EDIM + col);
      else            v = *(const float4*)(r2e + (size_t)sPR[row] * EDIM + (col - EDIM));
    }
    _Float16* d = sX + base;
    d[0] = (_Float16)v.x; d[1] = (_Float16)v.y;
    d[2] = (_Float16)v.z; d[3] = (_Float16)v.w;
  }
  __syncthreads();

  // x = relu([u,r] @ W1^T + b1)            (sX -> sY)   16 WMMA/wave
  wmma_layer<128>(sX, 128, w1h, w_1_b, sY, 64, wave, lane);
  __syncthreads();
  // o = relu(x @ W2^T + b2)                (sY -> sO)    8 WMMA/wave
  wmma_layer<64>(sY, 64, w2h, w_2_b, sO, 64, wave, lane);
  __syncthreads();
  // a = relu([o, post] @ A1^T + b)         (sO -> sY)   16 WMMA/wave
  wmma_att1(sO, sPostH, a1h, att1_b, sY, wave, lane);
  __syncthreads();
  // a = relu(a @ A2^T + b)                 (sY -> sY, per-wave rows: safe)
  wmma_layer<64>(sY, 64, a2h, att2_b, sY, 64, wave, lane);
  __syncthreads();

  // scores[l] = a[l] . att3 + b3
  if (tid < 64) {
    float acc = att3_b[0];
#pragma unroll
    for (int k = 0; k < EDIM; ++k) acc += (float)sY[tid * 64 + k] * att3_w[k];
    sScore[tid] = acc;
  }
  __syncthreads();

  // masked softmax over valid history (hl <= 50; serial cost trivial)
  if (tid == 0) {
    int hl = hist_len[n];
    if (hl > N_L) hl = N_L;
    float mx = -3.402823466e38f;
    for (int l = 0; l < hl; ++l) mx = fmaxf(mx, sScore[l]);
    float sum = 0.f;
    for (int l = 0; l < hl; ++l) { float e = __expf(sScore[l] - mx); sWt[l] = e; sum += e; }
    const float inv = 1.0f / sum;
    for (int l = 0; l < hl; ++l) sWt[l] *= inv;
    for (int l = hl; l < 64; ++l) sWt[l] = 0.f;
  }
  __syncthreads();

  // att_history[e] = sum_l w[l] * o[l][e]
  if (tid < EDIM) {
    float acc = 0.f;
    for (int l = 0; l < N_L; ++l) acc += sWt[l] * (float)sO[l * 64 + tid];
    sAH[tid] = acc;
  }
  __syncthreads();

  // out = relu([att_history, post_rep] @ Ow^T + b)
  if (tid < EDIM) {
    const float* wrow = o_w_w + (size_t)tid * CEDIM;
    float acc = o_w_b[tid];
#pragma unroll
    for (int k = 0; k < EDIM; ++k) acc += sAH[k] * wrow[k];
#pragma unroll
    for (int k = 0; k < EDIM; ++k) acc += sPost[k] * wrow[EDIM + k];
    out[(size_t)n * EDIM + tid] = fmaxf(acc, 0.0f);
  }
}

// ---------------------------------------------------------------------------
extern "C" void kernel_launch(void* const* d_in, const int* in_sizes, int n_in,
                              void* d_out, int out_size, void* d_ws, size_t ws_size,
                              hipStream_t stream) {
  const int*   nodes      = (const int*)d_in[0];
  const int*   pu_history = (const int*)d_in[1];
  const int*   pr_history = (const int*)d_in[2];
  const int*   hist_len   = (const int*)d_in[3];
  const float* u2e        = (const float*)d_in[4];
  const float* r2e        = (const float*)d_in[5];
  const float* p2e        = (const float*)d_in[6];
  const float* w_e_w      = (const float*)d_in[7];
  const float* w_e_b      = (const float*)d_in[8];
  const float* w_1_w      = (const float*)d_in[9];
  const float* w_1_b      = (const float*)d_in[10];
  const float* w_2_w      = (const float*)d_in[11];
  const float* w_2_b      = (const float*)d_in[12];
  const float* att1_w     = (const float*)d_in[13];
  const float* att1_b     = (const float*)d_in[14];
  const float* att2_w     = (const float*)d_in[15];
  const float* att2_b     = (const float*)d_in[16];
  const float* att3_w     = (const float*)d_in[17];
  const float* att3_b     = (const float*)d_in[18];
  const float* o_w_w      = (const float*)d_in[19];
  const float* o_w_b      = (const float*)d_in[20];
  float*       out        = (float*)d_out;

  const int N = in_sizes[0];  // number of nodes

  // f16 weight staging in workspace (total 48 KB)
  _Float16* ws  = (_Float16*)d_ws;
  _Float16* w1h = ws;                   // [64][128]
  _Float16* w2h = w1h + 64 * 128;       // [64][64]
  _Float16* a1h = w2h + 64 * 64;        // [64][128]
  _Float16* a2h = a1h + 64 * 128;       // [64][64]

  cvt_f32_to_f16<<<(64 * 128 + 255) / 256, 256, 0, stream>>>(w_1_w,  w1h, 64 * 128);
  cvt_f32_to_f16<<<(64 * 64  + 255) / 256, 256, 0, stream>>>(w_2_w,  w2h, 64 * 64);
  cvt_f32_to_f16<<<(64 * 128 + 255) / 256, 256, 0, stream>>>(att1_w, a1h, 64 * 128);
  cvt_f32_to_f16<<<(64 * 64  + 255) / 256, 256, 0, stream>>>(att2_w, a2h, 64 * 64);

  post_encode_kernel<<<N, 128, 0, stream>>>(
      nodes, pu_history, pr_history, hist_len, u2e, r2e, p2e,
      w_e_w, w_e_b, w1h, w_1_b, w2h, w_2_b,
      a1h, att1_b, a2h, att2_b, att3_w, att3_b, o_w_w, o_w_b, out);
}
